// ScaledDotProductAttention_6167573037296
// MI455X (gfx1250) — compile-verified, matrix operand order checked by reading, not run on hardware
//
#include <hip/hip_runtime.h>

#define B_    8
#define S_    1024
#define DM    1024
#define HEAD_ 16
#define DKV   64

typedef __bf16 bf16x16 __attribute__((ext_vector_type(16)));
typedef float  f32x8   __attribute__((ext_vector_type(8)));

union Frag16 {
    bf16x16        v;
    unsigned short u[16];
    uint4          q2[2];
};

union U4 {
    uint4          q;
    unsigned short u[8];
};

__device__ __forceinline__ unsigned short f2bf(float f) {
    unsigned u = __float_as_uint(f);
    u += 0x7FFFu + ((u >> 16) & 1u);   // round-to-nearest-even
    return (unsigned short)(u >> 16);
}

__device__ __forceinline__ unsigned pk2(float lo, float hi) {
    return (unsigned)f2bf(lo) | ((unsigned)f2bf(hi) << 16);
}

// ---------------------------------------------------------------------------
// Kernel 1: C[m,n] = sum_k X[m,k]*W[n,k] + bias[n]; bf16 out in [b,h,s,dkv].
// 128 threads = 4 waves. Workgroup tile 128x64; wave owns 32 rows x 64 cols
// (2 A-frags reused over 4 B-frags => 8 WMMAs per 12 b128 LDS loads).
// ---------------------------------------------------------------------------
__global__ __launch_bounds__(128)
void qkv_gemm_bf16(const float* __restrict__ X, const float* __restrict__ W,
                   const float* __restrict__ bias,
                   unsigned short* __restrict__ outbf) {
    __shared__ __align__(16) unsigned short Xs[128][40];  // 80B rows (16B-mult)
    __shared__ __align__(16) unsigned short Wsh[64][40];

    const unsigned t    = threadIdx.x;
    const unsigned lane = t & 31, wave = t >> 5;
    const unsigned half = (lane >> 4) & 1, l16 = lane & 15;
    const unsigned m0 = blockIdx.x * 128, n0 = blockIdx.y * 64;

    const f32x8 zero = {0.f,0.f,0.f,0.f,0.f,0.f,0.f,0.f};
    f32x8 acc[2][4];
#pragma unroll
    for (unsigned mi = 0; mi < 2; ++mi)
#pragma unroll
        for (unsigned tt = 0; tt < 4; ++tt) acc[mi][tt] = zero;

    for (unsigned k0 = 0; k0 < DM; k0 += 32) {
        __syncthreads();
        // X tile 128x32 f32: 8 float4 per thread -> pack -> ds_store_b64
#pragma unroll
        for (unsigned j = 0; j < 8; ++j) {
            unsigned i4 = t + 128u * j;
            unsigned r = i4 >> 3, c = (i4 & 7u) * 4u;
            float4 xv = *(const float4*)&X[(size_t)(m0 + r) * DM + k0 + c];
            uint2 pk; pk.x = pk2(xv.x, xv.y); pk.y = pk2(xv.z, xv.w);
            *(uint2*)&Xs[r][c] = pk;
        }
        // W tile 64x32 f32: 4 float4 per thread
#pragma unroll
        for (unsigned j = 0; j < 4; ++j) {
            unsigned i4 = t + 128u * j;
            unsigned r = i4 >> 3, c = (i4 & 7u) * 4u;
            float4 wv = *(const float4*)&W[(size_t)(n0 + r) * DM + k0 + c];
            uint2 pk; pk.x = pk2(wv.x, wv.y); pk.y = pk2(wv.z, wv.w);
            *(uint2*)&Wsh[r][c] = pk;
        }
        __syncthreads();

        Frag16 a[2];
#pragma unroll
        for (unsigned mi = 0; mi < 2; ++mi) {
            const unsigned short* row = &Xs[wave * 32 + mi * 16 + l16][0];
            a[mi].q2[0] = *(const uint4*)&row[half * 8];
            a[mi].q2[1] = *(const uint4*)&row[16 + half * 8];
        }
#pragma unroll
        for (unsigned tt = 0; tt < 4; ++tt) {
            Frag16 b;
            const unsigned short* row = &Wsh[tt * 16 + l16][0];
            b.q2[0] = *(const uint4*)&row[half * 8];
            b.q2[1] = *(const uint4*)&row[16 + half * 8];
#pragma unroll
            for (unsigned mi = 0; mi < 2; ++mi)
                acc[mi][tt] = __builtin_amdgcn_wmma_f32_16x16x32_bf16(
                    false, a[mi].v, false, b.v, (short)0, acc[mi][tt], false, false);
        }
    }

    // epilogue: bias add, bf16 convert, scatter into [b,h,s,dkv]
#pragma unroll
    for (unsigned tt = 0; tt < 4; ++tt) {
        unsigned n  = n0 + tt * 16 + l16;
        float    bv = bias[n];
        unsigned h = n >> 6, d = n & 63;
#pragma unroll
        for (unsigned mi = 0; mi < 2; ++mi) {
#pragma unroll
            for (unsigned r = 0; r < 8; ++r) {
                unsigned m  = m0 + wave * 32 + mi * 16 + r + 8 * half;
                unsigned bb = m >> 10, s = m & 1023;
                outbf[(((size_t)bb * HEAD_ + h) * S_ + s) * DKV + d] =
                    f2bf(acc[mi][tt][r] + bv);
            }
        }
    }
}

// ---------------------------------------------------------------------------
// Kernel 2: flash attention with post-softmax group_prob scaling.
// One workgroup per (b,h,q-block of 64). 128 threads = 4 waves; wave w owns
// q rows [16w,16w+16). l tracks sum(exp) (denominator, no group_prob);
// acc tracks sum(exp*g*v). V staged TRANSPOSED so its B-fragments are
// contiguous b128 LDS loads like K's.
// ---------------------------------------------------------------------------
__global__ __launch_bounds__(128)
void flash_attn_group(const unsigned short* __restrict__ Qb,
                      const unsigned short* __restrict__ Kb,
                      const unsigned short* __restrict__ Vb,
                      const float* __restrict__ G,
                      float* __restrict__ out) {
    __shared__ __align__(16) unsigned short Qs[64][72];   // 144B rows
    __shared__ __align__(16) unsigned short Ks[64][72];
    __shared__ __align__(16) unsigned short Vt[64][72];   // transposed: [d][k]
    __shared__ __align__(16) unsigned short Ps[64][72];

    const unsigned t    = threadIdx.x;
    const unsigned lane = t & 31, wave = t >> 5;
    const unsigned half = (lane >> 4) & 1, l16 = lane & 15;
    const unsigned qb = blockIdx.x;            // q-block 0..15
    const unsigned bh = blockIdx.y;            // b*HEAD + h
    const unsigned b = bh >> 4, h = bh & 15;
    const size_t   head_base = (size_t)bh * S_ * DKV;

    // stage Q tile [64 x 64] bf16 with b128 loads/stores
#pragma unroll
    for (unsigned j = 0; j < 4; ++j) {
        unsigned i4 = t + 128u * j;
        unsigned r = i4 >> 3, c = (i4 & 7u) * 8u;
        *(uint4*)&Qs[r][c] =
            *(const uint4*)&Qb[head_base + (size_t)(qb * 64 + r) * DKV + c];
    }
    __syncthreads();

    Frag16 qa[2];
    {
        const unsigned short* row = &Qs[wave * 16 + l16][0];
        qa[0].q2[0] = *(const uint4*)&row[half * 8];
        qa[0].q2[1] = *(const uint4*)&row[16 + half * 8];
        qa[1].q2[0] = *(const uint4*)&row[32 + half * 8];
        qa[1].q2[1] = *(const uint4*)&row[48 + half * 8];
    }

    const f32x8 zero = {0.f,0.f,0.f,0.f,0.f,0.f,0.f,0.f};
    float m_r[8], l_r[8];
    f32x8 acc[4];
#pragma unroll
    for (unsigned r = 0; r < 8; ++r) { m_r[r] = -3.0e38f; l_r[r] = 0.f; }
#pragma unroll
    for (unsigned tt = 0; tt < 4; ++tt) acc[tt] = zero;

    for (unsigned kb = 0; kb < S_; kb += 64) {
        __syncthreads();
        // K: row-major b128 stage; V: transposed scatter (Vt[d][k])
#pragma unroll
        for (unsigned j = 0; j < 4; ++j) {
            unsigned i4 = t + 128u * j;
            unsigned r = i4 >> 3, c = (i4 & 7u) * 8u;
            *(uint4*)&Ks[r][c] =
                *(const uint4*)&Kb[head_base + (size_t)(kb + r) * DKV + c];
            U4 vv;
            vv.q = *(const uint4*)&Vb[head_base + (size_t)(kb + r) * DKV + c];
#pragma unroll
            for (unsigned e = 0; e < 8; ++e) Vt[c + e][r] = vv.u[e];
        }
        // prefetch next K/V tiles (global_prefetch_b8) while this tile computes
        if (kb + 64 < S_) {
            const unsigned short* nk = &Kb[head_base + (size_t)(kb + 64) * DKV];
            const unsigned short* nv = &Vb[head_base + (size_t)(kb + 64) * DKV];
            __builtin_prefetch(nk + (size_t)t * 32, 0, 1);
            __builtin_prefetch(nv + (size_t)t * 32, 0, 1);
        }
        __syncthreads();

        // scores S = (Q K^T) / 8 for this wave's 16 q-rows x 64 k-cols
        f32x8 sc[4];
#pragma unroll
        for (unsigned tt = 0; tt < 4; ++tt) {
            f32x8 c0 = zero;
            const unsigned short* krow = &Ks[tt * 16 + l16][0];
            Frag16 b0, b1;     // column n of B=K^T is K-row n (contiguous)
            b0.q2[0] = *(const uint4*)&krow[half * 8];
            b0.q2[1] = *(const uint4*)&krow[16 + half * 8];
            b1.q2[0] = *(const uint4*)&krow[32 + half * 8];
            b1.q2[1] = *(const uint4*)&krow[48 + half * 8];
            c0 = __builtin_amdgcn_wmma_f32_16x16x32_bf16(
                false, qa[0].v, false, b0.v, (short)0, c0, false, false);
            c0 = __builtin_amdgcn_wmma_f32_16x16x32_bf16(
                false, qa[1].v, false, b1.v, (short)0, c0, false, false);
            sc[tt] = c0 * 0.125f;
        }

        // online softmax; rows live in 16-lane halves -> xor 1/2/4/8 reduces a row
        float scale[8];
#pragma unroll
        for (unsigned r = 0; r < 8; ++r) {
            float rm = fmaxf(fmaxf(sc[0][r], sc[1][r]), fmaxf(sc[2][r], sc[3][r]));
            rm = fmaxf(rm, __shfl_xor(rm, 1, 32));
            rm = fmaxf(rm, __shfl_xor(rm, 2, 32));
            rm = fmaxf(rm, __shfl_xor(rm, 4, 32));
            rm = fmaxf(rm, __shfl_xor(rm, 8, 32));
            float mn = fmaxf(m_r[r], rm);
            scale[r] = __expf(m_r[r] - mn);
            m_r[r] = mn;
            l_r[r] *= scale[r];
        }
#pragma unroll
        for (unsigned tt = 0; tt < 4; ++tt)
#pragma unroll
            for (unsigned r = 0; r < 8; ++r) acc[tt][r] *= scale[r];

        // e = exp(s-m): denominator uses e; numerator uses e*group_prob
        const unsigned qrow_base = qb * 64 + wave * 16;
        float esum[8];
#pragma unroll
        for (unsigned r = 0; r < 8; ++r) esum[r] = 0.f;
#pragma unroll
        for (unsigned tt = 0; tt < 4; ++tt) {
            unsigned kg = kb + tt * 16 + l16;
#pragma unroll
            for (unsigned r = 0; r < 8; ++r) {
                float e = __expf(sc[tt][r] - m_r[r]);
                esum[r] += e;
                unsigned qg = qrow_base + r + 8 * half;
                float g = G[((size_t)b * S_ + qg) * S_ + kg];
                Ps[wave * 16 + r + 8 * half][tt * 16 + l16] = f2bf(e * g);
            }
        }
#pragma unroll
        for (unsigned r = 0; r < 8; ++r) {
            float es = esum[r];
            es += __shfl_xor(es, 1, 32);
            es += __shfl_xor(es, 2, 32);
            es += __shfl_xor(es, 4, 32);
            es += __shfl_xor(es, 8, 32);
            l_r[r] += es;
        }
        __syncthreads();   // Ps C-layout -> A-layout reshuffle through LDS

        // acc += P @ V : V B-frags are contiguous b128 loads from Vt
        Frag16 pa0, pa1;
        {
            const unsigned short* prow = &Ps[wave * 16 + l16][0];
            pa0.q2[0] = *(const uint4*)&prow[half * 8];
            pa0.q2[1] = *(const uint4*)&prow[16 + half * 8];
            pa1.q2[0] = *(const uint4*)&prow[32 + half * 8];
            pa1.q2[1] = *(const uint4*)&prow[48 + half * 8];
        }
#pragma unroll
        for (unsigned tt = 0; tt < 4; ++tt) {
            const unsigned short* vrow = &Vt[tt * 16 + l16][0];
            Frag16 vb0, vb1;
            vb0.q2[0] = *(const uint4*)&vrow[half * 8];
            vb0.q2[1] = *(const uint4*)&vrow[16 + half * 8];
            vb1.q2[0] = *(const uint4*)&vrow[32 + half * 8];
            vb1.q2[1] = *(const uint4*)&vrow[48 + half * 8];
            acc[tt] = __builtin_amdgcn_wmma_f32_16x16x32_bf16(
                false, pa0.v, false, vb0.v, (short)0, acc[tt], false, false);
            acc[tt] = __builtin_amdgcn_wmma_f32_16x16x32_bf16(
                false, pa1.v, false, vb1.v, (short)0, acc[tt], false, false);
        }
    }

    // finalize: out[b, q, h*64 + d] = acc / l
#pragma unroll
    for (unsigned r = 0; r < 8; ++r) {
        float inv = 1.0f / l_r[r];
        unsigned qg = qb * 64 + wave * 16 + r + 8 * half;
#pragma unroll
        for (unsigned tt = 0; tt < 4; ++tt) {
            out[((size_t)b * S_ + qg) * (HEAD_ * DKV) + h * 64 + tt * 16 + l16] =
                acc[tt][r] * inv;
        }
    }
}

extern "C" void kernel_launch(void* const* d_in, const int* in_sizes, int n_in,
                              void* d_out, int out_size, void* d_ws, size_t ws_size,
                              hipStream_t stream) {
    const float* queries = (const float*)d_in[0];
    const float* keys    = (const float*)d_in[1];
    const float* values  = (const float*)d_in[2];
    const float* gprob   = (const float*)d_in[3];
    // d_in[4] attention_mask: intentionally unused (reference discards mask result)
    const float* Wq = (const float*)d_in[5];
    const float* bq = (const float*)d_in[6];
    const float* Wk = (const float*)d_in[7];
    const float* bk = (const float*)d_in[8];
    const float* Wv = (const float*)d_in[9];
    const float* bv = (const float*)d_in[10];

    unsigned short* ws   = (unsigned short*)d_ws;
    const size_t    per  = (size_t)B_ * HEAD_ * S_ * DKV;  // 8,388,608 bf16 elems
    unsigned short* qbuf = ws;
    unsigned short* kbuf = ws + per;
    unsigned short* vbuf = ws + 2 * per;

    dim3 blk(128);
    dim3 g1(B_ * S_ / 128, DM / 64);  // 64 x 16 tiles
    qkv_gemm_bf16<<<g1, blk, 0, stream>>>(queries, Wq, bq, qbuf);
    qkv_gemm_bf16<<<g1, blk, 0, stream>>>(keys,    Wk, bk, kbuf);
    qkv_gemm_bf16<<<g1, blk, 0, stream>>>(values,  Wv, bv, vbuf);

    dim3 g2(S_ / 64, B_ * HEAD_);     // 16 q-blocks x 128 (b,h)
    flash_attn_group<<<g2, blk, 0, stream>>>(qbuf, kbuf, vbuf, gprob, (float*)d_out);
}
